// GTS_Model_52518860095761
// MI455X (gfx1250) — compile-verified, matrix operand order checked by reading, not run on hardware
//
#include <hip/hip_runtime.h>
#include <hip/hip_bf16.h>
#include <math.h>

typedef _Float16 h16;
typedef __attribute__((ext_vector_type(16))) _Float16 v16h;
typedef __attribute__((ext_vector_type(8)))  _Float16 v8h;
typedef __attribute__((ext_vector_type(8)))  float    v8f;

#define GN 1000
#define GT 4000
#define GB 16
#define GS 12
#define GH 12
#define GHID 64
#define GF 65            // 1 + HID (valid features)
#define FP 96            // padded feature dim (3 k-steps of 32)
#define GD 128
#define GC1 8
#define GC2 16
#define GK 10
#define MP 1024          // padded node-rows (32 tiles of 32)
#define KP 1024          // padded node-K   (32 k-steps)
#define CATC (GB * FP)   // 1536 columns of A@cat

// ---------------------------------------------------------------------------
__global__ void zero_kernel(float* p, int n) {
    int i = blockIdx.x * blockDim.x + threadIdx.x;
    if (i < n) p[i] = 0.0f;
}

// in [Kin, Nin] f32 row-major -> out [Nin, Kp] f16 row-major (K-padded zeros)
__global__ void transpose_pad_kernel(const float* __restrict__ in, h16* __restrict__ out,
                                     int Kin, int Nin, int Kp) {
    int idx = blockIdx.x * blockDim.x + threadIdx.x;
    if (idx >= Nin * Kp) return;
    int n = idx / Kp;
    int k = idx % Kp;
    out[idx] = (h16)((k < Kin) ? in[k * Nin + n] : 0.0f);
}

// ---------------------------------------------------------------------------
// conv1 -> relu -> conv2 -> relu -> mean-pool -> fc -> relu   (per node)
// ---------------------------------------------------------------------------
__global__ void conv_feat_kernel(const float* __restrict__ x,      // [N, T]
                                 const float* __restrict__ w1,     // [8,1,10]
                                 const float* __restrict__ b1,     // [8]
                                 const float* __restrict__ w2,     // [16,8,10]
                                 const float* __restrict__ b2,     // [16]
                                 const float* __restrict__ fcw,    // [16,128]
                                 const float* __restrict__ fcb,    // [128]
                                 h16* __restrict__ f) {            // [MP,128] f16
    const int TS = 256;
    const int Tout1 = GT - GK + 1;          // 3991
    const int Tout2 = Tout1 - GK + 1;       // 3982
    __shared__ float sx[TS + 18];
    __shared__ float sc1[GC1 * (TS + 9)];
    __shared__ float sred[256];
    __shared__ float spool[GC2];

    const int n   = blockIdx.x;
    const int tid = threadIdx.x;
    const float* xn = x + n * GT;

    float pacc[GC2];
#pragma unroll
    for (int c = 0; c < GC2; ++c) pacc[c] = 0.0f;

    for (int t0 = 0; t0 < Tout2; t0 += TS) {
        for (int idx = tid; idx < TS + 18; idx += 256) {
            int t = t0 + idx;
            sx[idx] = (t < GT) ? xn[t] : 0.0f;
        }
        __syncthreads();
        for (int idx = tid; idx < GC1 * (TS + 9); idx += 256) {
            int c = idx / (TS + 9);
            int u = idx % (TS + 9);
            float acc = 0.0f;
            if (t0 + u < Tout1) {
                acc = b1[c];
#pragma unroll
                for (int k = 0; k < GK; ++k) acc += sx[u + k] * w1[c * GK + k];
                acc = fmaxf(acc, 0.0f);
            }
            sc1[c * (TS + 9) + u] = acc;
        }
        __syncthreads();
        int t = t0 + tid;
        if (t < Tout2) {
            for (int c2 = 0; c2 < GC2; ++c2) {
                float acc = b2[c2];
#pragma unroll
                for (int c1 = 0; c1 < GC1; ++c1) {
                    const float* wrow = w2 + (c2 * GC1 + c1) * GK;
                    const float* srow = sc1 + c1 * (TS + 9) + tid;
#pragma unroll
                    for (int k = 0; k < GK; ++k) acc += srow[k] * wrow[k];
                }
                pacc[c2] += fmaxf(acc, 0.0f);
            }
        }
        __syncthreads();
    }

    for (int c2 = 0; c2 < GC2; ++c2) {
        sred[tid] = pacc[c2];
        __syncthreads();
        for (int s = 128; s > 0; s >>= 1) {
            if (tid < s) sred[tid] += sred[tid + s];
            __syncthreads();
        }
        if (tid == 0) spool[c2] = sred[0] / (float)Tout2;
        __syncthreads();
    }
    if (tid < GD) {
        float acc = fcb[tid];
#pragma unroll
        for (int c2 = 0; c2 < GC2; ++c2) acc += spool[c2] * fcw[c2 * GD + tid];
        f[n * GD + tid] = (h16)fmaxf(acc, 0.0f);
    }
}

// ---------------------------------------------------------------------------
// f16 fragment load: P row-major f16, contiguous along K -> 2x global_load_b128
// K split per ISA 16-bit layout: lanes 0-15 take K {0..7,16..23}, lanes 16-31
// take {8..15,24..31}.
// ---------------------------------------------------------------------------
__device__ __forceinline__ v16h load_frag_h(const h16* __restrict__ P, int ld,
                                            int row, int k0, int klo) {
    const h16* base = P + row * ld + k0 + klo;
    v8h lo = *(const v8h*)(base);
    v8h hi = *(const v8h*)(base + 16);
    v16h v;
#pragma unroll
    for (int e = 0; e < 8; ++e) { v[e] = lo[e]; v[e + 8] = hi[e]; }
    return v;
}

__device__ __forceinline__ float apply_act(float v, int act) {
    if (act == 1) v = fmaxf(v, 0.0f);
    else if (act == 2) v = 1.0f / (1.0f + __expf(-v));
    else if (act == 3) v = tanhf(v);
    return v;
}

#define WMMA_F16(a, b, c) \
    __builtin_amdgcn_wmma_f32_16x16x32_f16(false, (a), false, (b), (short)0, (c), false, false)

// ---------------------------------------------------------------------------
// C = act( A1*B1 [+ A2*B2] + bias ).  A row-major f16 [M,K]; B TRANSPOSED
// row-major f16 [Ncol,K]; C row-major (f32 to Cf, or f16 to Ch if non-null).
// Requires M%32==0, Ncol%32==0, K%32==0 (caller pads).  32x32 tile per wave:
// 4 WMMAs per 4 fragment loads.
// ---------------------------------------------------------------------------
__global__ void wmma_gemm_h(const h16* __restrict__ A1, const h16* __restrict__ B1T,
                            const h16* __restrict__ A2, const h16* __restrict__ B2T,
                            const float* __restrict__ bias,
                            float* __restrict__ Cf, h16* __restrict__ Ch,
                            int M, int Ncol, int K, int act) {
    int tilesN = Ncol >> 5;
    int tiles  = (M >> 5) * tilesN;
    int wave   = blockIdx.x * (blockDim.x >> 5) + (threadIdx.x >> 5);
    if (wave >= tiles) return;
    int lane    = threadIdx.x & 31;
    int rowBase = (wave / tilesN) << 5;
    int colBase = (wave % tilesN) << 5;

    int ar0 = rowBase + (lane & 15);
    int ar1 = ar0 + 16;
    int bc0 = colBase + (lane & 15);
    int bc1 = bc0 + 16;
    int klo = (lane < 16) ? 0 : 8;

    v8f acc00 = {}, acc01 = {}, acc10 = {}, acc11 = {};

    if (A2) {
        for (int k0 = 0; k0 < K; k0 += 32) {
            v16h a0 = load_frag_h(A1, K, ar0, k0, klo);
            v16h a1 = load_frag_h(A1, K, ar1, k0, klo);
            v16h b0 = load_frag_h(B1T, K, bc0, k0, klo);
            v16h b1 = load_frag_h(B1T, K, bc1, k0, klo);
            acc00 = WMMA_F16(a0, b0, acc00);
            acc01 = WMMA_F16(a0, b1, acc01);
            acc10 = WMMA_F16(a1, b0, acc10);
            acc11 = WMMA_F16(a1, b1, acc11);
            v16h c0 = load_frag_h(A2, K, ar0, k0, klo);
            v16h c1 = load_frag_h(A2, K, ar1, k0, klo);
            v16h d0 = load_frag_h(B2T, K, bc0, k0, klo);
            v16h d1 = load_frag_h(B2T, K, bc1, k0, klo);
            acc00 = WMMA_F16(c0, d0, acc00);
            acc01 = WMMA_F16(c0, d1, acc01);
            acc10 = WMMA_F16(c1, d0, acc10);
            acc11 = WMMA_F16(c1, d1, acc11);
        }
    } else {
        for (int k0 = 0; k0 < K; k0 += 32) {
            // software prefetch of the next k-slab (global_prefetch_b8)
            __builtin_prefetch(A1 + ar0 * K + k0 + 32, 0, 1);
            __builtin_prefetch(B1T + bc0 * K + k0 + 32, 0, 1);
            v16h a0 = load_frag_h(A1, K, ar0, k0, klo);
            v16h a1 = load_frag_h(A1, K, ar1, k0, klo);
            v16h b0 = load_frag_h(B1T, K, bc0, k0, klo);
            v16h b1 = load_frag_h(B1T, K, bc1, k0, klo);
            acc00 = WMMA_F16(a0, b0, acc00);
            acc01 = WMMA_F16(a0, b1, acc01);
            acc10 = WMMA_F16(a1, b0, acc10);
            acc11 = WMMA_F16(a1, b1, acc11);
        }
    }

    // C/D layout: VGPR r -> row (lanes 0-15: r, lanes 16-31: 8+r), col = lane&15
    int nOut  = colBase + (lane & 15);
    int mBase = rowBase + ((lane < 16) ? 0 : 8);
    float bv0 = bias ? bias[nOut] : 0.0f;
    float bv1 = bias ? bias[nOut + 16] : 0.0f;
    if (Ch) {
#pragma unroll
        for (int r = 0; r < 8; ++r) {
            Ch[(mBase + r) * Ncol + nOut]           = (h16)apply_act(acc00[r] + bv0, act);
            Ch[(mBase + r) * Ncol + nOut + 16]      = (h16)apply_act(acc01[r] + bv1, act);
            Ch[(mBase + 16 + r) * Ncol + nOut]      = (h16)apply_act(acc10[r] + bv0, act);
            Ch[(mBase + 16 + r) * Ncol + nOut + 16] = (h16)apply_act(acc11[r] + bv1, act);
        }
    } else {
#pragma unroll
        for (int r = 0; r < 8; ++r) {
            Cf[(mBase + r) * Ncol + nOut]           = apply_act(acc00[r] + bv0, act);
            Cf[(mBase + r) * Ncol + nOut + 16]      = apply_act(acc01[r] + bv1, act);
            Cf[(mBase + 16 + r) * Ncol + nOut]      = apply_act(acc10[r] + bv0, act);
            Cf[(mBase + 16 + r) * Ncol + nOut + 16] = apply_act(acc11[r] + bv1, act);
        }
    }
}

// ---------------------------------------------------------------------------
// pairwise edge MLP + hard gumbel pick; one wave per (i,j) pair
// ---------------------------------------------------------------------------
__global__ void edge_mask_kernel(const float* __restrict__ fa, const float* __restrict__ fb,
                                 const float* __restrict__ eb1, const float* __restrict__ W2,
                                 const float* __restrict__ b2, const float* __restrict__ un,
                                 float* __restrict__ mask, float* __restrict__ rowsum) {
    int pair = blockIdx.x * (blockDim.x >> 5) + (threadIdx.x >> 5);
    if (pair >= GN * GN) return;
    int lane = threadIdx.x & 31;
    int i = pair / GN;
    int j = pair % GN;

    float a0 = 0.0f, a1 = 0.0f;
#pragma unroll
    for (int c = 0; c < 4; ++c) {
        int d = lane + 32 * c;
        float he = fmaxf(fa[i * GD + d] + fb[j * GD + d] + eb1[d], 0.0f);
        a0 += he * W2[d * 2 + 0];
        a1 += he * W2[d * 2 + 1];
    }
#pragma unroll
    for (int off = 16; off > 0; off >>= 1) {
        a0 += __shfl_xor(a0, off, 32);
        a1 += __shfl_xor(a1, off, 32);
    }
    if (lane == 0) {
        float u0 = un[pair * 2 + 0], u1 = un[pair * 2 + 1];
        float g0 = -logf(-logf(u0 + 1e-10f) + 1e-10f);
        float g1 = -logf(-logf(u1 + 1e-10f) + 1e-10f);
        float l0 = a0 + b2[0] + g0;
        float l1 = a1 + b2[1] + g1;
        float m  = (l0 >= l1) ? 1.0f : 0.0f;
        if (i == j) m = 0.0f;
        mask[pair] = m;
        if (m > 0.0f) atomicAdd(&rowsum[i], m);
    }
}

// Amat_h [MP,KP] f16, pre-zeroed; fill valid block row-normalized
__global__ void build_A_kernel(const float* __restrict__ mask,
                               const float* __restrict__ rowsum,
                               h16* __restrict__ A) {
    int idx = blockIdx.x * blockDim.x + threadIdx.x;
    if (idx >= GN * GN) return;
    int i = idx / GN;
    int j = idx % GN;
    A[i * KP + j] = (h16)(mask[idx] / (rowsum[i] + 1e-6f));
}

// ---------------------------------------------------------------------------
// cat written in BOTH f16 layouts:
//   cat_rm[(n*B+b)*FP + f]   (A-operand of weight GEMMs, flat [16000,96])
//   catT [(b*FP+f)*KP + n]   (B-operand of A@cat, contiguous along K)
// ---------------------------------------------------------------------------
__global__ void build_cat_kernel(const float* __restrict__ xptr, int xsB, int xsN,
                                 const float* __restrict__ h, const float* __restrict__ ru,
                                 h16* __restrict__ cat_rm, h16* __restrict__ catT) {
    int idx = blockIdx.x * blockDim.x + threadIdx.x;
    if (idx >= GN * GB * GF) return;
    int nb = idx / GF;
    int fd = idx % GF;
    int n = nb / GB, b = nb % GB;
    float v;
    if (fd == 0) {
        v = xptr[b * xsB + n * xsN];
    } else {
        int d = fd - 1;
        v = h[nb * GHID + d];
        if (ru) v *= ru[nb * (2 * GHID) + d];   // r gate (already sigmoid'ed)
    }
    h16 hv = (h16)v;
    cat_rm[nb * FP + fd] = hv;
    catT[(b * FP + fd) * KP + n] = hv;
}

__global__ void gru_update_kernel(float* __restrict__ h, const float* __restrict__ ru,
                                  const float* __restrict__ c) {
    int idx = blockIdx.x * blockDim.x + threadIdx.x;
    if (idx >= GN * GB * GHID) return;
    int nb = idx / GHID;
    int d  = idx % GHID;
    float u = ru[nb * (2 * GHID) + GHID + d];
    h[idx] = u * h[idx] + (1.0f - u) * c[idx];
}

__global__ void proj_out_kernel(const float* __restrict__ h, const float* __restrict__ pw,
                                const float* __restrict__ pb, float* __restrict__ yprev,
                                float* __restrict__ out, int t) {
    int nb = blockIdx.x * blockDim.x + threadIdx.x;
    if (nb >= GN * GB) return;
    int n = nb / GB, b = nb % GB;
    float acc = pb[0];
#pragma unroll
    for (int d = 0; d < GHID; ++d) acc += h[nb * GHID + d] * pw[d];
    yprev[nb] = acc;
    out[b * (GH * GN) + t * GN + n] = acc;
}

__global__ void loss_kernel(const float* __restrict__ out, const float* __restrict__ tgt,
                            float* __restrict__ acc, int n) {
    __shared__ float sred[256];
    int tid = threadIdx.x;
    float s = 0.0f;
    for (int i = blockIdx.x * blockDim.x + tid; i < n; i += gridDim.x * blockDim.x) {
        float d = out[i] - tgt[i];
        s += d * d;
    }
    sred[tid] = s;
    __syncthreads();
    for (int st = 128; st > 0; st >>= 1) {
        if (tid < st) sred[tid] += sred[tid + st];
        __syncthreads();
    }
    if (tid == 0) atomicAdd(acc, sred[0]);
}

__global__ void finalize_loss_kernel(const float* __restrict__ acc, float* __restrict__ loss, int n) {
    if (threadIdx.x == 0 && blockIdx.x == 0) loss[0] = acc[0] / (float)n;
}

// ---------------------------------------------------------------------------
// host side
// ---------------------------------------------------------------------------
static inline void launch_gemm(const h16* A1, const h16* B1T,
                               const h16* A2, const h16* B2T,
                               const float* bias, float* Cf, h16* Ch,
                               int M, int Ncol, int K, int act, hipStream_t s) {
    int tiles  = (M / 32) * (Ncol / 32);
    int blocks = (tiles + 7) / 8;   // 8 waves / 256-thread block
    wmma_gemm_h<<<blocks, 256, 0, s>>>(A1, B1T, A2, B2T, bias, Cf, Ch, M, Ncol, K, act);
}

static inline void launch_zero(void* p, size_t nbytes, hipStream_t s) {
    int n = (int)(nbytes / 4);
    zero_kernel<<<(n + 255) / 256, 256, 0, s>>>((float*)p, n);
}

static inline void launch_tpad(const float* in, h16* out, int Kin, int Nin, int Kp,
                               hipStream_t s) {
    transpose_pad_kernel<<<(Nin * Kp + 255) / 256, 256, 0, s>>>(in, out, Kin, Nin, Kp);
}

extern "C" void kernel_launch(void* const* d_in, const int* in_sizes, int n_in,
                              void* d_out, int out_size, void* d_ws, size_t ws_size,
                              hipStream_t stream) {
    const float* inputs   = (const float*)d_in[0];
    const float* entire   = (const float*)d_in[1];
    const float* targets  = (const float*)d_in[2];
    const float* u_noise  = (const float*)d_in[3];
    const float* conv1_w  = (const float*)d_in[4];
    const float* conv1_b  = (const float*)d_in[5];
    const float* conv2_w  = (const float*)d_in[6];
    const float* conv2_b  = (const float*)d_in[7];
    const float* fc_w     = (const float*)d_in[8];
    const float* fc_b     = (const float*)d_in[9];
    const float* Wa       = (const float*)d_in[10];
    const float* Wb       = (const float*)d_in[11];
    const float* b1       = (const float*)d_in[12];
    const float* W2       = (const float*)d_in[13];
    const float* b2       = (const float*)d_in[14];
    const float* eWru0    = (const float*)d_in[15];
    const float* eWru1    = (const float*)d_in[16];
    const float* ebru     = (const float*)d_in[17];
    const float* eWc0     = (const float*)d_in[18];
    const float* eWc1     = (const float*)d_in[19];
    const float* ebc      = (const float*)d_in[20];
    const float* dWru0    = (const float*)d_in[21];
    const float* dWru1    = (const float*)d_in[22];
    const float* dbru     = (const float*)d_in[23];
    const float* dWc0     = (const float*)d_in[24];
    const float* dWc1     = (const float*)d_in[25];
    const float* dbc      = (const float*)d_in[26];
    const float* proj_w   = (const float*)d_in[27];
    const float* proj_b   = (const float*)d_in[28];

    float* out   = (float*)d_out;
    float* maskO = out;                                  // [N*N]
    float* outsO = out + GN * GN;                        // [B*H*N]
    float* lossO = out + GN * GN + GB * GH * GN;         // [1]

    // ---- workspace layout (byte cursor, 64B aligned slots) ----
    char* base = (char*)d_ws;
    size_t o = 0;
    auto alloc = [&](size_t bytes) -> void* {
        void* p = base + o;
        o += (bytes + 63) & ~(size_t)63;
        return p;
    };
    h16*  f_h      = (h16*)  alloc((size_t)MP * GD * 2);
    float* fa      = (float*)alloc((size_t)MP * GD * 4);
    float* fb      = (float*)alloc((size_t)MP * GD * 4);
    float* rowsum  = (float*)alloc(4096);
    h16*  Amat_h   = (h16*)  alloc((size_t)MP * KP * 2);
    h16*  catT_h   = (h16*)  alloc((size_t)CATC * KP * 2);
    h16*  cat2T_h  = (h16*)  alloc((size_t)CATC * KP * 2);
    h16*  cat_rm   = (h16*)  alloc((size_t)GN * GB * FP * 2);
    h16*  cat2_rm  = (h16*)  alloc((size_t)GN * GB * FP * 2);
    h16*  agg_h    = (h16*)  alloc((size_t)MP * CATC * 2);
    h16*  agg2_h   = (h16*)  alloc((size_t)MP * CATC * 2);
    float* ru      = (float*)alloc((size_t)GN * GB * 2 * GHID * 4);
    float* cval    = (float*)alloc((size_t)GN * GB * GHID * 4);
    float* h       = (float*)alloc((size_t)GN * GB * GHID * 4);
    float* yprev   = (float*)alloc((size_t)GN * GB * 4);
    float* lacc    = (float*)alloc(64);
    h16*  WaT      = (h16*)  alloc((size_t)GD * GD * 2);
    h16*  WbT      = (h16*)  alloc((size_t)GD * GD * 2);
    h16*  eWru0T   = (h16*)  alloc((size_t)2 * GHID * FP * 2);
    h16*  eWru1T   = (h16*)  alloc((size_t)2 * GHID * FP * 2);
    h16*  eWc0T    = (h16*)  alloc((size_t)GHID * FP * 2);
    h16*  eWc1T    = (h16*)  alloc((size_t)GHID * FP * 2);
    h16*  dWru0T   = (h16*)  alloc((size_t)2 * GHID * FP * 2);
    h16*  dWru1T   = (h16*)  alloc((size_t)2 * GHID * FP * 2);
    h16*  dWc0T    = (h16*)  alloc((size_t)GHID * FP * 2);
    h16*  dWc1T    = (h16*)  alloc((size_t)GHID * FP * 2);

    const int NBF  = GN * GB * GF;    // cat build work items
    const int NBH  = GN * GB * GHID;  // 1,024,000
    const int NB   = GN * GB;         // 16,000
    const int OUTN = GB * GH * GN;    // 192,000

    // ---- fresh state every call (graph-replay safe; pads stay zero) ----
    launch_zero(f_h,     (size_t)MP * GD * 2, stream);
    launch_zero(rowsum,  (size_t)GN * 4, stream);
    launch_zero(Amat_h,  (size_t)MP * KP * 2, stream);
    launch_zero(catT_h,  (size_t)CATC * KP * 2, stream);
    launch_zero(cat2T_h, (size_t)CATC * KP * 2, stream);
    launch_zero(cat_rm,  (size_t)GN * GB * FP * 2, stream);
    launch_zero(cat2_rm, (size_t)GN * GB * FP * 2, stream);
    launch_zero(h,       (size_t)NBH * 4, stream);
    launch_zero(yprev,   (size_t)NB * 4, stream);
    launch_zero(lacc,    4, stream);

    // ---- pre-transpose / K-pad / f16-convert all GEMM B-operands ----
    launch_tpad(Wa,    WaT,    GD, GD,       GD, stream);
    launch_tpad(Wb,    WbT,    GD, GD,       GD, stream);
    launch_tpad(eWru0, eWru0T, GF, 2 * GHID, FP, stream);
    launch_tpad(eWru1, eWru1T, GF, 2 * GHID, FP, stream);
    launch_tpad(eWc0,  eWc0T,  GF, GHID,     FP, stream);
    launch_tpad(eWc1,  eWc1T,  GF, GHID,     FP, stream);
    launch_tpad(dWru0, dWru0T, GF, 2 * GHID, FP, stream);
    launch_tpad(dWru1, dWru1T, GF, 2 * GHID, FP, stream);
    launch_tpad(dWc0,  dWc0T,  GF, GHID,     FP, stream);
    launch_tpad(dWc1,  dWc1T,  GF, GHID,     FP, stream);

    // ---- graph learning ----
    conv_feat_kernel<<<GN, 256, 0, stream>>>(entire, conv1_w, conv1_b, conv2_w, conv2_b,
                                             fc_w, fc_b, f_h);
    launch_gemm(f_h, WaT, nullptr, nullptr, nullptr, fa, nullptr, MP, GD, GD, 0, stream);
    launch_gemm(f_h, WbT, nullptr, nullptr, nullptr, fb, nullptr, MP, GD, GD, 0, stream);
    edge_mask_kernel<<<(GN * GN + 7) / 8, 256, 0, stream>>>(fa, fb, b1, W2, b2, u_noise,
                                                            maskO, rowsum);
    build_A_kernel<<<(GN * GN + 255) / 256, 256, 0, stream>>>(maskO, rowsum, Amat_h);

    // ---- encoder scan ----
    for (int s = 0; s < GS; ++s) {
        const float* xs = inputs + s * GN;   // inputs[b,s,n,0] = xs[b*S*N + n]
        build_cat_kernel<<<(NBF + 255) / 256, 256, 0, stream>>>(xs, GS * GN, 1, h, nullptr,
                                                                cat_rm, catT_h);
        launch_gemm(Amat_h, catT_h, nullptr, nullptr, nullptr, nullptr, agg_h,
                    MP, CATC, KP, 0, stream);
        launch_gemm(cat_rm, eWru0T, agg_h, eWru1T, ebru, ru, nullptr,
                    NB, 2 * GHID, FP, 2, stream);
        build_cat_kernel<<<(NBF + 255) / 256, 256, 0, stream>>>(xs, GS * GN, 1, h, ru,
                                                                cat2_rm, cat2T_h);
        launch_gemm(Amat_h, cat2T_h, nullptr, nullptr, nullptr, nullptr, agg2_h,
                    MP, CATC, KP, 0, stream);
        launch_gemm(cat2_rm, eWc0T, agg2_h, eWc1T, ebc, cval, nullptr,
                    NB, GHID, FP, 3, stream);
        gru_update_kernel<<<(NBH + 255) / 256, 256, 0, stream>>>(h, ru, cval);
    }

    // ---- decoder scan (autoregressive) ----
    for (int t = 0; t < GH; ++t) {
        build_cat_kernel<<<(NBF + 255) / 256, 256, 0, stream>>>(yprev, 1, GB, h, nullptr,
                                                                cat_rm, catT_h);
        launch_gemm(Amat_h, catT_h, nullptr, nullptr, nullptr, nullptr, agg_h,
                    MP, CATC, KP, 0, stream);
        launch_gemm(cat_rm, dWru0T, agg_h, dWru1T, dbru, ru, nullptr,
                    NB, 2 * GHID, FP, 2, stream);
        build_cat_kernel<<<(NBF + 255) / 256, 256, 0, stream>>>(yprev, 1, GB, h, ru,
                                                                cat2_rm, cat2T_h);
        launch_gemm(Amat_h, cat2T_h, nullptr, nullptr, nullptr, nullptr, agg2_h,
                    MP, CATC, KP, 0, stream);
        launch_gemm(cat2_rm, dWc0T, agg2_h, dWc1T, dbc, cval, nullptr,
                    NB, GHID, FP, 3, stream);
        gru_update_kernel<<<(NBH + 255) / 256, 256, 0, stream>>>(h, ru, cval);
        proj_out_kernel<<<(NB + 255) / 256, 256, 0, stream>>>(h, proj_w, proj_b, yprev, outsO, t);
    }

    // ---- loss ----
    loss_kernel<<<256, 256, 0, stream>>>(outsO, targets, lacc, OUTN);
    finalize_loss_kernel<<<1, 32, 0, stream>>>(lacc, lossO, OUTN);
}
